// ImageCaptionModel_42047729828279
// MI455X (gfx1250) — compile-verified, matrix operand order checked by reading, not run on hardware
//
#include <hip/hip_runtime.h>
#include <hip/hip_bf16.h>

// ---------- types ----------
typedef __bf16 bf16_t;
typedef __attribute__((ext_vector_type(8)))  __bf16 bf16x8;
typedef __attribute__((ext_vector_type(16))) __bf16 bf16x16;
typedef __attribute__((ext_vector_type(8)))  float  f32x8;
typedef __attribute__((ext_vector_type(4)))  int    v4i;

// Problem constants
#define BB   128      // batch
#define TT   25       // time steps
#define RR   100      // attention regions
#define DD   512      // feature / hidden dim
#define VV   10000    // vocab
#define HH   512
#define G4   2048     // 4*H gates

// CDNA5 async global->LDS path (ASYNCcnt), guarded so a missing builtin
// falls back to the load+ds_store path.
#if defined(__gfx1250__) && __has_builtin(__builtin_amdgcn_global_load_async_to_lds_b128) && __has_builtin(__builtin_amdgcn_s_wait_asynccnt)
#define USE_ASYNC_LDS 1
#else
#define USE_ASYNC_LDS 0
#endif

__device__ __forceinline__ float sigf(float x) { return 1.0f / (1.0f + __expf(-x)); }

#if USE_ASYNC_LDS
// Builtin signature (from toolchain diagnostic): (v4i AS1*, v4i AS3*, Ii offset, Ii cpol).
// Build the AS pointers via integer casts: AS1 uses the full 64-bit VA; AS3 uses the
// low 32 bits of the generic pointer, which per the flat-aperture rule IS the LDS offset.
typedef __attribute__((address_space(1))) v4i* gv4i_p;
typedef __attribute__((address_space(3))) v4i* lv4i_p;

__device__ __forceinline__ void async_b128(const bf16_t* gsrc, bf16_t* ldst)
{
    gv4i_p g = (gv4i_p)(unsigned long long)(uintptr_t)gsrc;
    lv4i_p l = (lv4i_p)(unsigned long long)(unsigned)(uintptr_t)ldst;
    __builtin_amdgcn_global_load_async_to_lds_b128(g, l, 0, 0);
}
#endif

// ======================================================================
// WMMA bf16 GEMM:  C[M x Nout](f32) = A[M x K](bf16, row lda)
//                                   x Bt[Npad x K](bf16, pre-transposed)
//                  (+ bias[N])  (+ addmat[M x N], row ldadd)
// Block = 256 thr = 8 waves. Block tile: 128(M) x 64(N). K step 32.
// B panel double-buffered in LDS via async global->LDS DMA.
// gridDim.x = Npad/64, gridDim.y = M/128.
// ======================================================================
__global__ void __launch_bounds__(256)
wmma_gemm_bf16(const bf16_t* __restrict__ A, int lda,
               const bf16_t* __restrict__ Bt,
               const float* __restrict__ bias,
               const float* __restrict__ addmat, int ldadd,
               float* __restrict__ C, int ldc,
               int K, int Nout)
{
    __shared__ bf16_t ldsB[2][64 * 40];  // 2 x (64 N-rows x 32 K, padded to 40)

    const int wave = threadIdx.x >> 5;
    const int lane = threadIdx.x & 31;
    const int hf   = lane >> 4;          // wave half: selects K sub-block per ISA layout
    const int lrow = lane & 15;
    const int nbase = blockIdx.x * 64;
    const int mbase = blockIdx.y * 128;

    // staging assignment: 256 threads x 16B covers the 64x32 bf16 panel
    const int sr  = threadIdx.x >> 2;        // 0..63  (N row)
    const int scx = (threadIdx.x & 3) * 8;   // 0,8,16,24 (K col)
    const bf16_t* Bpan = Bt + (size_t)(nbase + sr) * K + scx;

    f32x8 acc[4] = {};                   // four 16x16 f32 accumulators
    const bf16_t* Arow = A + (size_t)(mbase + wave * 16 + lrow) * lda;
    const int nsteps = K >> 5;

#if USE_ASYNC_LDS
    // ---- prologue: DMA panel 0 into buffer 0 ----
    async_b128(Bpan, &ldsB[0][sr * 40 + scx]);
    __builtin_amdgcn_s_wait_asynccnt(0);
    __syncthreads();

    for (int i = 0; i < nsteps; ++i) {
        const int k0  = i << 5;
        const int cur = i & 1;
        if (i + 1 < nsteps)                     // DMA next panel while computing
            async_b128(Bpan + k0 + 32, &ldsB[cur ^ 1][sr * 40 + scx]);

        // A fragment (global; A panel is L2/L1 resident)
        bf16x8 alo = *(const bf16x8*)(Arow + k0 + 8 * hf);
        bf16x8 ahi = *(const bf16x8*)(Arow + k0 + 16 + 8 * hf);
        bf16x16 afrag = __builtin_shufflevector(alo, ahi,
            0,1,2,3,4,5,6,7,8,9,10,11,12,13,14,15);

        // load all four B fragments first, then issue WMMAs back-to-back
        bf16x16 bfrag[4];
        #pragma unroll
        for (int j = 0; j < 4; ++j) {
            const bf16_t* brow = &ldsB[cur][(j * 16 + lrow) * 40];
            bf16x8 blo = *(const bf16x8*)(brow + 8 * hf);
            bf16x8 bhi = *(const bf16x8*)(brow + 16 + 8 * hf);
            bfrag[j] = __builtin_shufflevector(blo, bhi,
                0,1,2,3,4,5,6,7,8,9,10,11,12,13,14,15);
        }
        #pragma unroll
        for (int j = 0; j < 4; ++j)
            acc[j] = __builtin_amdgcn_wmma_f32_16x16x32_bf16(
                false, afrag, false, bfrag[j], (short)0, acc[j], false, false);

        if (i + 1 < nsteps)
            __builtin_amdgcn_s_wait_asynccnt(0);  // my panel DMA done
        __syncthreads();                          // everyone done reading cur / writing nxt
    }
#else
    // ---- fallback: load + ds_store staging, two barriers per step ----
    for (int i = 0; i < nsteps; ++i) {
        const int k0 = i << 5;
        {
            const bf16x8 v = *(const bf16x8*)(Bpan + k0);
            *(bf16x8*)(&ldsB[0][sr * 40 + scx]) = v;
            if (k0 + 32 < K)
                __builtin_prefetch(Bpan + k0 + 32, 0, 1);
        }
        __syncthreads();
        bf16x8 alo = *(const bf16x8*)(Arow + k0 + 8 * hf);
        bf16x8 ahi = *(const bf16x8*)(Arow + k0 + 16 + 8 * hf);
        bf16x16 afrag = __builtin_shufflevector(alo, ahi,
            0,1,2,3,4,5,6,7,8,9,10,11,12,13,14,15);
        bf16x16 bfrag[4];
        #pragma unroll
        for (int j = 0; j < 4; ++j) {
            const bf16_t* brow = &ldsB[0][(j * 16 + lrow) * 40];
            bf16x8 blo = *(const bf16x8*)(brow + 8 * hf);
            bf16x8 bhi = *(const bf16x8*)(brow + 16 + 8 * hf);
            bfrag[j] = __builtin_shufflevector(blo, bhi,
                0,1,2,3,4,5,6,7,8,9,10,11,12,13,14,15);
        }
        #pragma unroll
        for (int j = 0; j < 4; ++j)
            acc[j] = __builtin_amdgcn_wmma_f32_16x16x32_bf16(
                false, afrag, false, bfrag[j], (short)0, acc[j], false, false);
        __syncthreads();
    }
#endif

    // --- store D: VGPR v holds M = v + 8*hf ; lane lrow holds N column ---
    #pragma unroll
    for (int j = 0; j < 4; ++j) {
        const int n = nbase + j * 16 + lrow;
        if (n >= Nout) continue;
        const float bv = bias ? bias[n] : 0.0f;
        #pragma unroll
        for (int v = 0; v < 8; ++v) {
            const int m = mbase + wave * 16 + v + 8 * hf;
            float val = acc[j][v] + bv;
            if (addmat) val += addmat[(size_t)m * ldadd + n];
            C[(size_t)m * ldc + n] = val;
        }
    }
}

// ======================================================================
// LSTM gate fusion: z[128 x 2048] = [i f o g], S[128 x 1024] = [h || c]
// updated in place; new h also emitted as bf16 to up to two destinations.
// ======================================================================
__global__ void lstm_gates(const float* __restrict__ z,
                           float* __restrict__ S,
                           bf16_t* __restrict__ dst0, int s0,
                           bf16_t* __restrict__ dst1, int s1)
{
    const int idx = blockIdx.x * blockDim.x + threadIdx.x;  // 128*512
    if (idx >= BB * HH) return;
    const int b = idx >> 9, j = idx & 511;
    const float zi = z[b * G4 + j];
    const float zf = z[b * G4 + 512 + j];
    const float zo = z[b * G4 + 1024 + j];
    const float zg = z[b * G4 + 1536 + j];
    const float c  = S[b * 1024 + 512 + j];
    const float cn = sigf(zf) * c + sigf(zi) * tanhf(zg);
    const float hn = sigf(zo) * tanhf(cn);
    S[b * 1024 + j]       = hn;
    S[b * 1024 + 512 + j] = cn;
    dst0[(size_t)b * s0 + j] = (bf16_t)hn;
    if (dst1) dst1[(size_t)b * s1 + j] = (bf16_t)hn;
}

// ======================================================================
// Attention: scores = S0 @ attn_W + attn_b, softmax over R, ctx = a @ feats
// One block (256 thr = 8 waves) per batch row.
// ======================================================================
__global__ void __launch_bounds__(256)
attn_ctx(const float* __restrict__ S0,       // 128 x 1024
         const float* __restrict__ attn_W,   // 1024 x 100
         const float* __restrict__ attn_b,   // 100
         const float* __restrict__ feats,    // 128 x 100 x 512
         bf16_t* __restrict__ ctx_dst, int ctx_stride)
{
    __shared__ float sc[RR];
    __shared__ float ew[RR];
    const int b  = blockIdx.x;
    const int wv = threadIdx.x >> 5;
    const int ln = threadIdx.x & 31;
    const float* s = S0 + (size_t)b * 1024;

    for (int r = wv; r < RR; r += 8) {
        float p = 0.0f;
        for (int k = ln; k < 1024; k += 32) p += s[k] * attn_W[k * RR + r];
        #pragma unroll
        for (int off = 16; off > 0; off >>= 1) p += __shfl_down(p, off, 32);
        if (ln == 0) sc[r] = p + attn_b[r];
    }
    __syncthreads();

    float mx = -3.0e38f;
    for (int r = 0; r < RR; ++r) mx = fmaxf(mx, sc[r]);
    float sum = 0.0f;
    for (int r = 0; r < RR; ++r) sum += __expf(sc[r] - mx);
    if (threadIdx.x < RR) ew[threadIdx.x] = __expf(sc[threadIdx.x] - mx) / sum;
    __syncthreads();

    for (int d = threadIdx.x; d < DD; d += 256) {
        float a = 0.0f;
        const float* fb = feats + ((size_t)b * RR) * DD + d;
        for (int r = 0; r < RR; ++r) a += ew[r] * fb[(size_t)r * DD];
        ctx_dst[(size_t)b * ctx_stride + d] = (bf16_t)a;
    }
}

// ======================================================================
// Prep: transpose+convert f32 weight block -> bf16 (N x K layout, padded)
// ======================================================================
__global__ void transpose_bf16(const float* __restrict__ src, int src_ld, int row0,
                               int Kd, int Nd, long total, bf16_t* __restrict__ dst)
{
    const long idx = (long)blockIdx.x * blockDim.x + threadIdx.x;
    if (idx >= total) return;
    const int n = (int)(idx / Kd), k = (int)(idx % Kd);
    const float v = (n < Nd) ? src[(size_t)(row0 + k) * src_ld + n] : 0.0f;
    dst[idx] = (bf16_t)v;
}

// Build XC (3200 x 1024 bf16): row (t*128+b) = [emb[tokens[b,t]] || maxpool10(feats[b])]
__global__ void build_xc(const int* __restrict__ tokens, const float* __restrict__ emb,
                         const float* __restrict__ feats, bf16_t* __restrict__ XC)
{
    const int idx = blockIdx.x * blockDim.x + threadIdx.x;
    if (idx >= TT * BB * 1024) return;
    const int r = idx >> 10, col = idx & 1023;
    const int t = r / BB, b = r % BB;
    float v;
    if (col < DD) {
        v = emb[(size_t)tokens[b * TT + t] * DD + col];
    } else {
        const int d = col - DD;
        v = feats[((size_t)b * RR) * DD + d];
        for (int rr = 1; rr < 10; ++rr)
            v = fmaxf(v, feats[((size_t)b * RR + rr) * DD + d]);
    }
    XC[idx] = (bf16_t)v;
}

// Init state buffers from initial_hidden_state (2 x 128 x 1024)
__global__ void init_state(const float* __restrict__ init,
                           float* __restrict__ S0, float* __restrict__ S1,
                           bf16_t* __restrict__ h0b, bf16_t* __restrict__ in1)
{
    const int idx = blockIdx.x * blockDim.x + threadIdx.x;
    if (idx >= BB * 1024) return;
    const float v0 = init[idx];
    const float v1 = init[BB * 1024 + idx];
    S0[idx] = v0; S1[idx] = v1;
    const int b = idx >> 10, j = idx & 1023;
    if (j < HH) {
        h0b[(size_t)b * HH + j]           = (bf16_t)v0;   // h0 for z0 GEMM
        in1[(size_t)b * 1536 + 1024 + j]  = (bf16_t)v1;   // h1 slot of LSTM1 input
    }
}

// Copy final hidden (2 x 128 x 1024) into tail of d_out
__global__ void finalize(const float* __restrict__ S0, const float* __restrict__ S1,
                         float* __restrict__ out_tail)
{
    const int idx = blockIdx.x * blockDim.x + threadIdx.x;
    if (idx >= 2 * BB * 1024) return;
    out_tail[idx] = (idx < BB * 1024) ? S0[idx] : S1[idx - BB * 1024];
}

// ======================================================================
extern "C" void kernel_launch(void* const* d_in, const int* in_sizes, int n_in,
                              void* d_out, int out_size, void* d_ws, size_t ws_size,
                              hipStream_t stream)
{
    (void)in_sizes; (void)n_in; (void)out_size; (void)ws_size;
    const int*   tokens = (const int*)  d_in[0];
    const float* feats  = (const float*)d_in[1];
    const float* init_h = (const float*)d_in[2];
    const float* emb    = (const float*)d_in[3];
    const float* W0     = (const float*)d_in[4];
    const float* b0     = (const float*)d_in[5];
    const float* W1     = (const float*)d_in[6];
    const float* b1     = (const float*)d_in[7];
    const float* attn_W = (const float*)d_in[8];
    const float* attn_b = (const float*)d_in[9];
    const float* out_W  = (const float*)d_in[10];
    const float* out_b  = (const float*)d_in[11];
    float* out = (float*)d_out;

    // ---- workspace carve-up ----
    char* p = (char*)d_ws;
    auto carve = [&](size_t bytes) { char* r = p; p += (bytes + 255) & ~(size_t)255; return r; };
    const int VPAD = 10048;                                   // 157 * 64
    bf16_t* W0a_t  = (bf16_t*)carve((size_t)G4 * 1024 * 2);   // W0[:1024]^T  (2048 x 1024)
    bf16_t* W0h_t  = (bf16_t*)carve((size_t)G4 * 512 * 2);    // W0[1024:]^T  (2048 x 512)
    bf16_t* W1_t   = (bf16_t*)carve((size_t)G4 * 1536 * 2);   // W1^T         (2048 x 1536)
    bf16_t* oW_t   = (bf16_t*)carve((size_t)VPAD * 512 * 2);  // out_W^T pad  (10048 x 512)
    bf16_t* XC     = (bf16_t*)carve((size_t)TT * BB * 1024 * 2);
    float*  xc0    = (float*) carve((size_t)TT * BB * G4 * 4);
    float*  zbuf   = (float*) carve((size_t)BB * G4 * 4);
    float*  S0     = (float*) carve((size_t)BB * 1024 * 4);
    float*  S1     = (float*) carve((size_t)BB * 1024 * 4);
    bf16_t* h0b    = (bf16_t*)carve((size_t)BB * 512 * 2);
    bf16_t* in1    = (bf16_t*)carve((size_t)BB * 1536 * 2);   // [h0new || ctx || h1]

    // ---- prep: transpose/convert weights to bf16 (N x K) ----
    {
        long tot;
        tot = (long)G4 * 1024;
        transpose_bf16<<<(tot + 255) / 256, 256, 0, stream>>>(W0, G4, 0, 1024, G4, tot, W0a_t);
        tot = (long)G4 * 512;
        transpose_bf16<<<(tot + 255) / 256, 256, 0, stream>>>(W0, G4, 1024, 512, G4, tot, W0h_t);
        tot = (long)G4 * 1536;
        transpose_bf16<<<(tot + 255) / 256, 256, 0, stream>>>(W1, G4, 0, 1536, G4, tot, W1_t);
        tot = (long)VPAD * 512;
        transpose_bf16<<<(tot + 255) / 256, 256, 0, stream>>>(out_W, VV, 0, 512, VV, tot, oW_t);
    }
    build_xc<<<(TT * BB * 1024 + 255) / 256, 256, 0, stream>>>(tokens, emb, feats, XC);
    init_state<<<(BB * 1024 + 255) / 256, 256, 0, stream>>>(init_h, S0, S1, h0b, in1);

    // ---- hoisted GEMM: xc0[t,b,:] = [x_t || pool] @ W0[:1024] + b0  (3200x1024x2048) ----
    wmma_gemm_bf16<<<dim3(G4 / 64, TT), 256, 0, stream>>>(
        XC, 1024, W0a_t, b0, nullptr, 0, xc0, G4, 1024, G4);

    // ---- recurrence ----
    for (int t = 0; t < TT; ++t) {
        // z0 = h0 @ W0[1024:1536] + xc0[t]          (128 x 512 x 2048)
        wmma_gemm_bf16<<<dim3(G4 / 64, 1), 256, 0, stream>>>(
            h0b, 512, W0h_t, nullptr, xc0 + (size_t)t * BB * G4, G4, zbuf, G4, 512, G4);
        // gates -> S0, h0 bf16 (for next z0) and in1[:, :512]
        lstm_gates<<<(BB * HH + 255) / 256, 256, 0, stream>>>(
            zbuf, S0, h0b, 512, in1, 1536);
        // attention context -> in1[:, 512:1024]
        attn_ctx<<<BB, 256, 0, stream>>>(S0, attn_W, attn_b, feats, in1 + 512, 1536);
        // z1 = [h0new || ctx || h1] @ W1 + b1       (128 x 1536 x 2048)
        wmma_gemm_bf16<<<dim3(G4 / 64, 1), 256, 0, stream>>>(
            in1, 1536, W1_t, b1, nullptr, 0, zbuf, G4, 1536, G4);
        // gates -> S1, h1 bf16 -> in1[:, 1024:1536] (doubles as logits A)
        lstm_gates<<<(BB * HH + 255) / 256, 256, 0, stream>>>(
            zbuf, S1, in1 + 1024, 1536, nullptr, 0);
        // logits[:, t, :] = h1new @ out_W + out_b   (128 x 512 x 10000)
        wmma_gemm_bf16<<<dim3(VPAD / 64, 1), 256, 0, stream>>>(
            in1 + 1024, 1536, oW_t, out_b, nullptr, 0,
            out + (size_t)t * VV, (size_t)TT * VV, 512, VV);
    }

    // ---- final hidden state ----
    finalize<<<(2 * BB * 1024 + 255) / 256, 256, 0, stream>>>(
        S0, S1, out + (size_t)BB * TT * VV);
}
